// DeepARFinancial_75737453298053
// MI455X (gfx1250) — compile-verified, matrix operand order checked by reading, not run on hardware
//
#include <hip/hip_runtime.h>
#include <hip/hip_fp16.h>
#include <math.h>

typedef __attribute__((ext_vector_type(16))) _Float16 v16h;
typedef __attribute__((ext_vector_type(8)))  float    v8f;

#define T_STEPS 8192
#define D_IN    128
#define H_DIM   1024
#define G4      (4 * H_DIM)          // 4096 gate rows
#define SCAN_WGS 64                  // persistent workgroups (grid barrier members)
#define SCAN_THREADS 256             // 8 waves of 32
#define ROWS_PER_WG (G4 / SCAN_WGS)  // 64 gate rows per WG
#define HIDX_PER_WG (H_DIM / SCAN_WGS) // 16 h-indices per WG
#define WROW_STRIDE 1028             // 1024 + 4 pad (16B aligned, bank-shifted)

// LDS floats: weights + h copy + z + c
#define SCAN_SMEM_FLOATS (ROWS_PER_WG * WROW_STRIDE + H_DIM + ROWS_PER_WG + HIDX_PER_WG)
#define SCAN_SMEM_BYTES  (SCAN_SMEM_FLOATS * 4)

// ---------------------------------------------------------------------------
// CDNA5 async global->LDS copy (ASYNCcnt-tracked, no VGPR round trip).
// ISA operand order: vdst = LDS byte address (low 32 bits of generic ptr),
//                    vaddr = 64-bit global address, saddr = off.
// ---------------------------------------------------------------------------
__device__ __forceinline__ void async_g2l_b128(void* lds_dst, const void* gsrc) {
  unsigned lds_off = (unsigned)(uintptr_t)lds_dst;
  unsigned long long ga = (unsigned long long)(uintptr_t)gsrc;
  asm volatile("global_load_async_to_lds_b128 %0, %1, off"
               :
               : "v"(lds_off), "v"(ga)
               : "memory");
}
__device__ __forceinline__ void wait_async0() {
  asm volatile("s_wait_asynccnt 0" ::: "memory");
}

// ---------------------------------------------------------------------------
// init: zero h double-buffers (2 layers x 2 buffers x 1024) and 2 barrier ctrs
// ---------------------------------------------------------------------------
__global__ void k_init(float* hbufs, unsigned* bars) {
  int i = threadIdx.x;
  for (int k = i; k < 4 * H_DIM; k += blockDim.x) hbufs[k] = 0.0f;
  if (i < 2) bars[i] = 0u;
}

// ---------------------------------------------------------------------------
// shifted obs -> f16 : row 0 zeros, row t = obs[t-1]  (flat shift by D)
// ---------------------------------------------------------------------------
__global__ void k_shift_obs_f16(const float* __restrict__ obs, _Float16* __restrict__ out) {
  int idx = blockIdx.x * blockDim.x + threadIdx.x;
  if (idx >= T_STEPS * D_IN) return;
  out[idx] = (idx < D_IN) ? (_Float16)0.0f : (_Float16)obs[idx - D_IN];
}

__global__ void k_f32_to_f16(const float* __restrict__ src, _Float16* __restrict__ dst, int n) {
  int i = blockIdx.x * blockDim.x + threadIdx.x;
  if (i < n) dst[i] = (_Float16)src[i];
}

// ---------------------------------------------------------------------------
// WMMA GEMM: C[M,N] = A[M,K] * B[N,K]^T + bias[N]
//   block = 256 threads = 8 waves; wave computes a 16x64 C strip (4 wmma acc).
//   mode 0: plain fp32 store.  mode 1: decoder epilogue -> loc / softplus scale
// ---------------------------------------------------------------------------
__launch_bounds__(256)
__global__ void k_gemm_wmma(const _Float16* __restrict__ A, const _Float16* __restrict__ B,
                            const float* __restrict__ bias, float* __restrict__ C,
                            int M, int N, int K, int mode) {
  const int lane = threadIdx.x & 31;
  const int wv   = threadIdx.x >> 5;
  const int m0   = (blockIdx.x * 8 + wv) * 16;
  const int n0   = blockIdx.y * 64;
  const bool hi  = lane >= 16;
  const int rsel = lane & 15;

  const _Float16* arow = A + (size_t)(m0 + rsel) * K;
  const _Float16* bp0 = B + (size_t)(n0 +  0 + rsel) * K;
  const _Float16* bp1 = B + (size_t)(n0 + 16 + rsel) * K;
  const _Float16* bp2 = B + (size_t)(n0 + 32 + rsel) * K;
  const _Float16* bp3 = B + (size_t)(n0 + 48 + rsel) * K;

  v8f acc0 = {}, acc1 = {}, acc2 = {}, acc3 = {};

  for (int k0 = 0; k0 < K; k0 += 32) {
    const int kb = k0 + (hi ? 16 : 0);
    if (k0 + 64 < K) {
      __builtin_prefetch(arow + kb + 64, 0, 3);   // near-temporal prefetch
      __builtin_prefetch(bp0 + kb + 64, 0, 3);
    }
    v16h av = *(const v16h*)(arow + kb);
    v16h b0 = *(const v16h*)(bp0 + kb);
    v16h b1 = *(const v16h*)(bp1 + kb);
    v16h b2 = *(const v16h*)(bp2 + kb);
    v16h b3 = *(const v16h*)(bp3 + kb);
    acc0 = __builtin_amdgcn_wmma_f32_16x16x32_f16(false, av, false, b0, (short)0, acc0, false, false);
    acc1 = __builtin_amdgcn_wmma_f32_16x16x32_f16(false, av, false, b1, (short)0, acc1, false, false);
    acc2 = __builtin_amdgcn_wmma_f32_16x16x32_f16(false, av, false, b2, (short)0, acc2, false, false);
    acc3 = __builtin_amdgcn_wmma_f32_16x16x32_f16(false, av, false, b3, (short)0, acc3, false, false);
  }

  // C/D layout: VGPR r, lanes 0-15 -> M=r, lanes 16-31 -> M=8+r; N = lane%16
  const int mrow = m0 + (hi ? 8 : 0);
  v8f accs[4] = {acc0, acc1, acc2, acc3};
#pragma unroll
  for (int j = 0; j < 4; ++j) {
    const int n = n0 + j * 16 + rsel;
    const float bv = bias[n];
#pragma unroll
    for (int r = 0; r < 8; ++r) {
      const float v = accs[j][r] + bv;
      const int m = mrow + r;
      if (mode == 0) {
        C[(size_t)m * N + n] = v;
      } else {
        if (n < D_IN) {
          C[(size_t)m * D_IN + n] = v;                       // loc
        } else {
          float sp = (v > 20.0f) ? v : log1pf(__expf(v));    // softplus
          C[(size_t)T_STEPS * D_IN + (size_t)m * D_IN + (n - D_IN)] = sp + 1.0e-4f;
        }
      }
    }
  }
}

// ---------------------------------------------------------------------------
// Persistent cooperative LSTM scan for one layer.
//   z = gx_all[t] + W_hh @ h_prev ; gates ; h,c update.
//   W_hh rows for this WG live in LDS (fp32) for all 8192 steps, staged with
//   async global->LDS DMA. Grid-wide split barrier: monotonic counter,
//   rel/acq device-scope atomics. h double-buffered in global so one barrier
//   per step suffices.
// ---------------------------------------------------------------------------
__launch_bounds__(SCAN_THREADS)
__global__ void k_lstm_scan(const float* __restrict__ gx_all, const float* __restrict__ Whh,
                            float* __restrict__ hbuf /* 2*H_DIM */,
                            _Float16* __restrict__ h_all, unsigned* __restrict__ bar) {
  extern __shared__ float smem[];
  float* wrows = smem;                               // ROWS_PER_WG * WROW_STRIDE
  float* h_lds = wrows + ROWS_PER_WG * WROW_STRIDE;  // H_DIM
  float* z_lds = h_lds + H_DIM;                      // ROWS_PER_WG
  float* c_lds = z_lds + ROWS_PER_WG;                // HIDX_PER_WG

  const int tid  = threadIdx.x;
  const int wg   = blockIdx.x;
  const int base = wg * HIDX_PER_WG;

  // ---- one-time weight staging into LDS via async DMA (16B per lane/row) ----
  for (int r = 0; r < ROWS_PER_WG; ++r) {
    const int grow = (r >> 4) * H_DIM + base + (r & 15);   // gate*H + h-index
    async_g2l_b128(wrows + r * WROW_STRIDE + tid * 4,
                   Whh + (size_t)grow * H_DIM + tid * 4);
  }
  if (tid < HIDX_PER_WG) c_lds[tid] = 0.0f;
  wait_async0();
  __syncthreads();

  const int wv = tid >> 5;
  const int lane = tid & 31;

  for (int t = 0; t < T_STEPS; ++t) {
    const float* hsrc = hbuf + (t & 1) * H_DIM;
    float*       hdst = hbuf + ((t + 1) & 1) * H_DIM;

    // async copy h_prev into LDS (256 lanes x 16B = 4KB)
    async_g2l_b128(h_lds + tid * 4, hsrc + tid * 4);
    wait_async0();
    __syncthreads();

    // 64 row-dots: wave wv owns rows [wv*8, wv*8+8); conflict-free float4 LDS reads
    for (int rr = 0; rr < 8; ++rr) {
      const int r = wv * 8 + rr;
      const float* wr = wrows + r * WROW_STRIDE;
      float p = 0.0f;
#pragma unroll
      for (int kk = 0; kk < 8; ++kk) {
        const int k = lane * 4 + kk * 128;
        float4 w4 = *(const float4*)(wr + k);
        float4 h4 = *(const float4*)(h_lds + k);
        p += w4.x * h4.x + w4.y * h4.y + w4.z * h4.z + w4.w * h4.w;
      }
      p += __shfl_down(p, 16, 32);
      p += __shfl_down(p,  8, 32);
      p += __shfl_down(p,  4, 32);
      p += __shfl_down(p,  2, 32);
      p += __shfl_down(p,  1, 32);
      if (lane == 0) z_lds[r] = p;
    }
    __syncthreads();

    // add precomputed gx (includes bias)
    if (tid < ROWS_PER_WG) {
      const int g = tid >> 4, i = tid & 15;
      z_lds[tid] += gx_all[(size_t)t * G4 + (size_t)g * H_DIM + base + i];
    }
    __syncthreads();

    // gates + state update for this WG's 16 h-indices
    if (tid < HIDX_PER_WG) {
      const float zi = z_lds[tid];
      const float zf = z_lds[16 + tid];
      const float zg = z_lds[32 + tid];
      const float zo = z_lds[48 + tid];
      const float ig = 1.0f / (1.0f + __expf(-zi));
      const float fg = 1.0f / (1.0f + __expf(-zf));
      const float gg = tanhf(zg);
      const float og = 1.0f / (1.0f + __expf(-zo));
      const float c = fg * c_lds[tid] + ig * gg;
      c_lds[tid] = c;
      const float h = og * tanhf(c);
      hdst[base + tid] = h;
      h_all[(size_t)t * H_DIM + base + tid] = (_Float16)h;
      __threadfence();   // make h visible at device scope before arrival
    }
    __syncthreads();

    // grid-wide split barrier (monotonic counter, no reset race)
    if (tid == 0) {
      __hip_atomic_fetch_add(bar, 1u, __ATOMIC_RELEASE, __HIP_MEMORY_SCOPE_AGENT);
      const unsigned target = (unsigned)(t + 1) * (unsigned)SCAN_WGS;
      while (__hip_atomic_load(bar, __ATOMIC_ACQUIRE, __HIP_MEMORY_SCOPE_AGENT) < target)
        __builtin_amdgcn_s_sleep(2);
    }
    __syncthreads();
    __threadfence();     // acquire-ish: invalidate near caches before reading new h
  }
}

// ---------------------------------------------------------------------------
// host-side orchestration
// ---------------------------------------------------------------------------
extern "C" void kernel_launch(void* const* d_in, const int* in_sizes, int n_in,
                              void* d_out, int out_size, void* d_ws, size_t ws_size,
                              hipStream_t stream) {
  (void)in_sizes; (void)n_in; (void)out_size; (void)ws_size;

  const float* obs   = (const float*)d_in[0];
  const float* W_ih0 = (const float*)d_in[1];
  const float* W_hh0 = (const float*)d_in[2];
  const float* b0    = (const float*)d_in[3];
  const float* W_ih1 = (const float*)d_in[4];
  const float* W_hh1 = (const float*)d_in[5];
  const float* b1    = (const float*)d_in[6];
  const float* W_dec = (const float*)d_in[7];
  const float* b_dec = (const float*)d_in[8];
  float* out = (float*)d_out;

  char* wsb = (char*)d_ws;
  size_t cur = 0;
  auto take = [&](size_t bytes) -> void* {
    void* p = wsb + cur;
    cur += (bytes + 255) & ~(size_t)255;
    return p;
  };

  // gx buffer reused by both layers (layer0 scan finishes before GEMM2 writes)
  float*    gx_buf = (float*)   take((size_t)T_STEPS * G4 * sizeof(float));      // 134 MB
  _Float16* h0f    = (_Float16*)take((size_t)T_STEPS * H_DIM * sizeof(_Float16));// 16.8 MB
  _Float16* h1f    = (_Float16*)take((size_t)T_STEPS * H_DIM * sizeof(_Float16));// 16.8 MB
  _Float16* obs_s  = (_Float16*)take((size_t)T_STEPS * D_IN * sizeof(_Float16));
  _Float16* wih0h  = (_Float16*)take((size_t)G4 * D_IN * sizeof(_Float16));
  _Float16* wih1h  = (_Float16*)take((size_t)G4 * H_DIM * sizeof(_Float16));
  _Float16* wdech  = (_Float16*)take((size_t)2 * D_IN * H_DIM * sizeof(_Float16));
  float*    hbufs  = (float*)   take((size_t)4 * H_DIM * sizeof(float)); // 2 layers x 2 bufs
  unsigned* bars   = (unsigned*)take(256);

  // 0) deterministic state init (ws is poisoned once, never re-poisoned)
  k_init<<<1, 256, 0, stream>>>(hbufs, bars);

  // 1) f16 conversions
  {
    int n = T_STEPS * D_IN;
    k_shift_obs_f16<<<(n + 255) / 256, 256, 0, stream>>>(obs, obs_s);
  }
  { int n = G4 * D_IN;        k_f32_to_f16<<<(n + 255) / 256, 256, 0, stream>>>(W_ih0, wih0h, n); }
  { int n = G4 * H_DIM;       k_f32_to_f16<<<(n + 255) / 256, 256, 0, stream>>>(W_ih1, wih1h, n); }
  { int n = 2 * D_IN * H_DIM; k_f32_to_f16<<<(n + 255) / 256, 256, 0, stream>>>(W_dec, wdech, n); }

  // 2) gx0_all = shifted_obs @ W_ih0^T + b0   (M=8192, N=4096, K=128)
  k_gemm_wmma<<<dim3(T_STEPS / 128, G4 / 64), 256, 0, stream>>>(
      obs_s, wih0h, b0, gx_buf, T_STEPS, G4, D_IN, 0);

  // 3) layer-0 scan (weights resident in LDS, grid barrier per step)
  k_lstm_scan<<<SCAN_WGS, SCAN_THREADS, SCAN_SMEM_BYTES, stream>>>(
      gx_buf, W_hh0, hbufs, h0f, bars + 0);

  // 4) gx1_all = h0_all @ W_ih1^T + b1   (M=8192, N=4096, K=1024) — reuse gx_buf
  k_gemm_wmma<<<dim3(T_STEPS / 128, G4 / 64), 256, 0, stream>>>(
      h0f, wih1h, b1, gx_buf, T_STEPS, G4, H_DIM, 0);

  // 5) layer-1 scan
  k_lstm_scan<<<SCAN_WGS, SCAN_THREADS, SCAN_SMEM_BYTES, stream>>>(
      gx_buf, W_hh1, hbufs + 2 * H_DIM, h1f, bars + 1);

  // 6) decoder: params = h1_all @ W_dec^T + b_dec, fused loc / softplus-scale
  k_gemm_wmma<<<dim3(T_STEPS / 128, (2 * D_IN) / 64), 256, 0, stream>>>(
      h1f, wdech, b_dec, out, T_STEPS, 2 * D_IN, H_DIM, 1);
}
